// EfficientAttentionBlock_34007551050480
// MI455X (gfx1250) — matrix operand order repaired and checked by reading, not compiled, not run on hardware
//
#include <hip/hip_runtime.h>

// ---------------------------------------------------------------------------
// EfficientAttentionBlock on MI455X (gfx1250, wave32, WMMA bf16 16x16x32)
// B=8, C=256, H=W=64, NH=4, HD=64, reduction=2 -> J=1024, I=4096
// GEMMs: double-buffered GLOBAL_LOAD_ASYNC_TO_LDS_B128 staging + WMMA from LDS.
// ---------------------------------------------------------------------------

#define BATCH 8
#define CCH   256
#define HWN   4096
#define NH    4
#define HD    64
#define JP    1024

typedef float v8f  __attribute__((ext_vector_type(8)));
typedef __bf16 bf16x16 __attribute__((ext_vector_type(16)));

union Frag16 { bf16x16 v; unsigned int u[8]; };

// A fragment: 16x32 bf16, row-major source (K contiguous pairs).
// lane&15 = M row, lane>>4 = half. VGPR v: K = 2*(v&3) + 16*(v>=4) + 8*half.
__device__ __forceinline__ void load_a_frag(Frag16& f, const __bf16* base,
                                            int row0, int ld, int k0, int lane) {
  const int h = lane >> 4, m = lane & 15;
  const __bf16* rowp = base + (size_t)(row0 + m) * ld + k0;
#pragma unroll
  for (int v = 0; v < 8; ++v) {
    const int kk = 2 * (v & 3) + ((v >= 4) ? 16 : 0) + h * 8;
    f.u[v] = *reinterpret_cast<const unsigned int*>(rowp + kk);
  }
}

// B fragment: 32x16 bf16. Source stored K-major: BT[n][k] with leading dim ldn.
// lane&15 = N, lane>>4 = half. VGPR v: K = 2*(v&3) + 16*(v>=4) + 8*half.
__device__ __forceinline__ void load_b_frag(Frag16& f, const __bf16* base,
                                            int n0, int ldn, int k0, int lane) {
  const int h = lane >> 4, n = lane & 15;
  const __bf16* rowp = base + (size_t)(n0 + n) * ldn + k0;
#pragma unroll
  for (int v = 0; v < 8; ++v) {
    const int kk = 2 * (v & 3) + ((v >= 4) ? 16 : 0) + h * 8;
    f.u[v] = *reinterpret_cast<const unsigned int*>(rowp + kk);
  }
}

__device__ __forceinline__ v8f wmma_bf16(const Frag16& a, const Frag16& b, v8f c) {
  return __builtin_amdgcn_wmma_f32_16x16x32_bf16(
      false, a.v, false, b.v, (short)0, c, false, false);
}

__device__ __forceinline__ v8f zero_v8f() {
  v8f z = {0.f, 0.f, 0.f, 0.f, 0.f, 0.f, 0.f, 0.f};
  return z;
}

// Async 16B copy global -> LDS (ASYNCcnt tracked, no VGPR staging).
// VDST = VGPR holding wave-relative LDS byte address; VADDR = 64-bit address.
__device__ __forceinline__ void async_copy16(void* lds, const void* gaddr) {
  unsigned lds_off = (unsigned)(unsigned long long)lds;
  asm volatile("global_load_async_to_lds_b128 %0, %1, off"
               :: "v"(lds_off), "v"(gaddr)
               : "memory");
}
__device__ __forceinline__ void wait_async_le3() {
  asm volatile("s_wait_asynccnt 0x3" ::: "memory");
}
__device__ __forceinline__ void wait_async_0() {
  asm volatile("s_wait_asynccnt 0x0" ::: "memory");
}

// ---------------------------------------------------------------------------
// Shared GEMM core: D[64 x 128] tile of  A[M x 256] * B^T  (B stored K-major,
// ldb = lda = 256).  8 waves = 4(M) x 2(N); each wave 16 x 64 -> 4 C frags.
// Double-buffered async-to-LDS staging of A(64x32) and B(128x32) tiles.
// ---------------------------------------------------------------------------
__device__ __forceinline__ void issue_tile(const __bf16* Ag, const __bf16* Bg,
                                           int m0b, int n0b, int k0, int tid,
                                           __bf16* As, __bf16* Bs) {
  {
    const int row = tid >> 2, seg = tid & 3;          // 64 rows x 4 chunks
    async_copy16(As + row * 32 + seg * 8,
                 Ag + (size_t)(m0b + row) * CCH + k0 + seg * 8);
  }
#pragma unroll
  for (int i = 0; i < 2; ++i) {                       // 128 rows x 4 chunks
    const int c = tid + i * 256;
    const int row = c >> 2, seg = c & 3;
    async_copy16(Bs + row * 32 + seg * 8,
                 Bg + (size_t)(n0b + row) * CCH + k0 + seg * 8);
  }
}

__device__ __forceinline__ void gemm64x128(const __bf16* Ag, const __bf16* Bg,
                                           int m0b, int n0b, int tid,
                                           v8f acc[4]) {
  __shared__ __align__(16) __bf16 As[2][64 * 32];
  __shared__ __align__(16) __bf16 Bs[2][128 * 32];

  const int wave = tid >> 5, lane = tid & 31;
  const int mrow = (wave >> 1) * 16;       // wave's M row inside 64-row tile
  const int ncol = (wave & 1) * 64;        // wave's N base inside 128-col tile

#pragma unroll
  for (int t = 0; t < 4; ++t) acc[t] = zero_v8f();

  issue_tile(Ag, Bg, m0b, n0b, 0, tid, As[0], Bs[0]);

  for (int k0 = 0; k0 < CCH; k0 += 32) {
    const int cur = (k0 >> 5) & 1;
    const bool has_next = (k0 + 32) < CCH;
    if (has_next) {
      issue_tile(Ag, Bg, m0b, n0b, k0 + 32, tid, As[cur ^ 1], Bs[cur ^ 1]);
      wait_async_le3();                    // current stage (3 older ops) done
    } else {
      wait_async_0();
    }
    __syncthreads();                       // tiles visible to all waves

    Frag16 a;
    load_a_frag(a, As[cur], mrow, 32, 0, lane);
#pragma unroll
    for (int t = 0; t < 4; ++t) {
      Frag16 bm;
      load_b_frag(bm, Bs[cur], ncol + t * 16, 32, 0, lane);
      acc[t] = wmma_bf16(a, bm, acc[t]);
    }
    __syncthreads();                       // done reading before overwrite
  }
}

// ---------------------------------------------------------------------------
// K1: per-(sample,chunk) partial sums for GroupNorm(1, C)
// ---------------------------------------------------------------------------
__global__ __launch_bounds__(256) void gn_partial(const float* __restrict__ x,
                                                  float2* __restrict__ partials) {
  const int b = blockIdx.x >> 6, chunk = blockIdx.x & 63;
  const float4* xp =
      reinterpret_cast<const float4*>(x + ((size_t)b << 20) + (size_t)chunk * 16384);
  float s = 0.f, q = 0.f;
#pragma unroll
  for (int it = 0; it < 16; ++it) {
    float4 v = xp[it * 256 + threadIdx.x];
    s += v.x + v.y + v.z + v.w;
    q += v.x * v.x + v.y * v.y + v.z * v.z + v.w * v.w;
  }
  for (int off = 16; off >= 1; off >>= 1) {
    s += __shfl_xor(s, off, 32);
    q += __shfl_xor(q, off, 32);
  }
  __shared__ float ss[8], qq[8];
  const int wave = threadIdx.x >> 5, lane = threadIdx.x & 31;
  if (lane == 0) { ss[wave] = s; qq[wave] = q; }
  __syncthreads();
  if (threadIdx.x == 0) {
    float S = 0.f, Q = 0.f;
    for (int w = 0; w < 8; ++w) { S += ss[w]; Q += qq[w]; }
    partials[blockIdx.x] = make_float2(S, Q);
  }
}

// K2: finalize mean/rstd per sample. grid 8 x 64
__global__ void gn_final(const float2* __restrict__ partials,
                         float2* __restrict__ stats) {
  const int b = blockIdx.x;
  float2 p = partials[b * 64 + threadIdx.x];
  float s = p.x, q = p.y;
  for (int off = 16; off >= 1; off >>= 1) {
    s += __shfl_xor(s, off, 32);
    q += __shfl_xor(q, off, 32);
  }
  __shared__ float ss[2], qq[2];
  const int wave = threadIdx.x >> 5;
  if ((threadIdx.x & 31) == 0) { ss[wave] = s; qq[wave] = q; }
  __syncthreads();
  if (threadIdx.x == 0) {
    const float inv_n = 1.0f / 1048576.0f;
    float S = ss[0] + ss[1], Q = qq[0] + qq[1];
    float mean = S * inv_n;
    float var = Q * inv_n - mean * mean;
    stats[b] = make_float2(mean, rsqrtf(var + 1e-5f));
  }
}

// K3: convert weights to bf16
__global__ void cvt_w(const float* __restrict__ wq, const float* __restrict__ wp,
                      __bf16* __restrict__ wqb, __bf16* __restrict__ wpb) {
  const int n1 = 3 * CCH * CCH, n2 = CCH * CCH;
  for (int i = blockIdx.x * 256 + threadIdx.x; i < n1 + n2; i += gridDim.x * 256) {
    if (i < n1) wqb[i] = (__bf16)wq[i];
    else        wpb[i - n1] = (__bf16)wp[i - n1];
  }
}

// K4: xn = gamma*(x-mu)*rstd + beta, written TRANSPOSED to xnT[b][hw][c] (bf16)
__global__ __launch_bounds__(256) void norm_xnT(const float* __restrict__ x,
                                                const float2* __restrict__ stats,
                                                const float* __restrict__ g,
                                                const float* __restrict__ be,
                                                __bf16* __restrict__ xnT) {
  const int b = blockIdx.x >> 4;
  const int hw = ((blockIdx.x & 15) << 8) + threadIdx.x;
  const float2 st = stats[b];
  const float* xb = x + (size_t)b * CCH * HWN;
  __bf16* o = xnT + ((size_t)b * HWN + hw) * CCH;
#pragma unroll 4
  for (int c = 0; c < CCH; ++c) {
    float v = (xb[(size_t)c * HWN + hw] - st.x) * st.y * g[c] + be[c];
    o[c] = (__bf16)v;
  }
}

// K5: QKV GEMM. D[768 x 4096] = Wqkv[768 x 256] x xn[256 x 4096] per sample.
__global__ __launch_bounds__(256) void qkv_gemm(const __bf16* __restrict__ A,
                                                const __bf16* __restrict__ xnT,
                                                __bf16* __restrict__ qT,
                                                __bf16* __restrict__ kf,
                                                __bf16* __restrict__ vf) {
  const int idx = blockIdx.x;
  const int b = idx / 384;
  const int rem = idx % 384;
  const int mt = rem >> 5;        // 0..11
  const int nt = rem & 31;        // 0..31
  const int tid = threadIdx.x, wave = tid >> 5, lane = tid & 31;
  const int h = lane >> 4, lr = lane & 15;
  const __bf16* Bbase = xnT + (size_t)b * HWN * CCH;

  v8f acc[4];
  gemm64x128(A, Bbase, mt * 64, nt * 128, tid, acc);

  const int m0 = mt * 64 + (wave >> 1) * 16;
  const int nbase = nt * 128 + (wave & 1) * 64;
#pragma unroll
  for (int t = 0; t < 4; ++t) {
#pragma unroll
    for (int r = 0; r < 8; ++r) {
      const int o = m0 + r + 8 * h;
      const int hw = nbase + t * 16 + lr;
      const float val = acc[t][r];
      if (o < 256) {
        const int head = o >> 6, cc = o & 63;
        qT[(((size_t)(b * NH + head)) * HWN + hw) * HD + cc] = (__bf16)val;
      } else if (o < 512) {
        kf[((size_t)b * CCH + (o - 256)) * HWN + hw] = (__bf16)val;
      } else {
        vf[((size_t)b * CCH + (o - 512)) * HWN + hw] = (__bf16)val;
      }
    }
  }
}

// K6: 2x2 average pool; kpT[b][head][j][c'] (c fastest), vp[b][head][c'][j].
__global__ void pool_kv(const __bf16* __restrict__ kf, const __bf16* __restrict__ vf,
                        __bf16* __restrict__ kpT, __bf16* __restrict__ vp) {
  const int total = BATCH * CCH * JP;
  for (int i = blockIdx.x * blockDim.x + threadIdx.x; i < total;
       i += gridDim.x * blockDim.x) {
    const int j = i & 1023;
    const int c = (i >> 10) & 255;
    const int b = i >> 18;
    const int jy = j >> 5, jx = j & 31;
    const size_t s0 = ((size_t)b * CCH + c) * HWN + (size_t)(jy * 2) * 64 + jx * 2;
    const float kv = 0.25f * ((float)kf[s0] + (float)kf[s0 + 1] +
                              (float)kf[s0 + 64] + (float)kf[s0 + 65]);
    const float vv = 0.25f * ((float)vf[s0] + (float)vf[s0 + 1] +
                              (float)vf[s0 + 64] + (float)vf[s0 + 65]);
    const int head = c >> 6, cc = c & 63;
    kpT[(((size_t)(b * NH + head)) * JP + j) * HD + cc] = (__bf16)kv;
    vp[(((size_t)(b * NH + head)) * HD + cc) * JP + j] = (__bf16)vv;
  }
}

// K7: fused attention. One block = (b, head, 16-row i-tile).
__global__ __launch_bounds__(256) void attn_kernel(const __bf16* __restrict__ qT,
                                                   const __bf16* __restrict__ kpT,
                                                   const __bf16* __restrict__ vp,
                                                   __bf16* __restrict__ outT) {
  __shared__ float wred[8][16];
  __shared__ float rowstat[16];
  __shared__ __align__(16) __bf16 Pbuf[16 * JP];
  __shared__ float outacc[16 * HD];

  const int idx = blockIdx.x;
  const int b = idx >> 10;
  const int head = (idx >> 8) & 3;
  const int i0 = (idx & 255) << 4;
  const int tid = threadIdx.x;
  const int wave = tid >> 5, lane = tid & 31;
  const int h = lane >> 4, lr = lane & 15;

  const __bf16* qbase = qT + ((size_t)(b * NH + head)) * HWN * HD;   // [i][c]
  const __bf16* kbase = kpT + ((size_t)(b * NH + head)) * JP * HD;   // [j][c]
  const __bf16* vbase = vp + ((size_t)(b * NH + head)) * HD * JP;    // [c][j]

  for (int e = tid; e < 16 * HD; e += 256) outacc[e] = 0.f;

  // ---- stage 1: logits --------------------------------------------------
  v8f s[8];
#pragma unroll
  for (int f = 0; f < 8; ++f) s[f] = zero_v8f();
  const int jw = wave * 128;
#pragma unroll
  for (int k0 = 0; k0 < HD; k0 += 32) {
    Frag16 a;
    load_a_frag(a, qbase, i0, HD, k0, lane);
#pragma unroll
    for (int f = 0; f < 8; ++f) {
      Frag16 bm;
      load_b_frag(bm, kbase, jw + f * 16, HD, k0, lane);
      s[f] = wmma_bf16(a, bm, s[f]);
    }
  }
  const float scale = 0.0625f;  // 256^-0.5
#pragma unroll
  for (int f = 0; f < 8; ++f)
#pragma unroll
    for (int r = 0; r < 8; ++r) s[f][r] *= scale;

  // row max: within-wave (8 frags + 16-lane group), then cross-wave via LDS
  float pm[8];
#pragma unroll
  for (int r = 0; r < 8; ++r) {
    float m = -1e30f;
#pragma unroll
    for (int f = 0; f < 8; ++f) m = fmaxf(m, s[f][r]);
    for (int off = 8; off >= 1; off >>= 1) m = fmaxf(m, __shfl_xor(m, off, 32));
    pm[r] = m;
  }
  if (lr == 0) {
#pragma unroll
    for (int r = 0; r < 8; ++r) wred[wave][h * 8 + r] = pm[r];
  }
  __syncthreads();                       // A (also covers outacc zeroing)
  if (tid < 16) {
    float m = wred[0][tid];
    for (int w = 1; w < 8; ++w) m = fmaxf(m, wred[w][tid]);
    rowstat[tid] = m;
  }
  __syncthreads();                       // B
  float rmax[8];
#pragma unroll
  for (int r = 0; r < 8; ++r) rmax[r] = rowstat[h * 8 + r];

  float ps[8];
#pragma unroll
  for (int r = 0; r < 8; ++r) {
    float acc = 0.f;
#pragma unroll
    for (int f = 0; f < 8; ++f) {
      float e = __expf(s[f][r] - rmax[r]);
      s[f][r] = e;
      acc += e;
    }
    for (int off = 8; off >= 1; off >>= 1) acc += __shfl_xor(acc, off, 32);
    ps[r] = acc;
  }
  if (lr == 0) {
#pragma unroll
    for (int r = 0; r < 8; ++r) wred[wave][h * 8 + r] = ps[r];
  }
  __syncthreads();                       // C
  if (tid < 16) {
    float ssum = 0.f;
    for (int w = 0; w < 8; ++w) ssum += wred[w][tid];
    rowstat[tid] = ssum;
  }
  __syncthreads();                       // D
#pragma unroll
  for (int r = 0; r < 8; ++r) {
    const float rinv = 1.0f / rowstat[h * 8 + r];
#pragma unroll
    for (int f = 0; f < 8; ++f)
      Pbuf[(r + 8 * h) * JP + jw + f * 16 + lr] = (__bf16)(s[f][r] * rinv);
  }
  __syncthreads();                       // E

  // ---- stage 2: out = P x v^T (K split over waves) ----------------------
  v8f acc2[4];
#pragma unroll
  for (int t = 0; t < 4; ++t) acc2[t] = zero_v8f();
  const int kw = wave * 128;
#pragma unroll
  for (int ks = 0; ks < 128; ks += 32) {
    Frag16 a;
    load_a_frag(a, (const __bf16*)Pbuf, 0, JP, kw + ks, lane);
#pragma unroll
    for (int t = 0; t < 4; ++t) {
      Frag16 bm;
      load_b_frag(bm, vbase, t * 16, JP, kw + ks, lane);
      acc2[t] = wmma_bf16(a, bm, acc2[t]);
    }
  }
#pragma unroll
  for (int t = 0; t < 4; ++t)
#pragma unroll
    for (int r = 0; r < 8; ++r)
      atomicAdd(&outacc[(r + 8 * h) * HD + t * 16 + lr], acc2[t][r]);
  __syncthreads();

  for (int e = tid; e < 16 * HD; e += 256) {
    const int m = e >> 6, c = e & 63;
    outT[((size_t)b * HWN + i0 + m) * CCH + head * HD + c] = (__bf16)outacc[e];
  }
}

// K8: proj GEMM + bias + residual (xn recomputed in f32). Output f32.
__global__ __launch_bounds__(256) void proj_kernel(const __bf16* __restrict__ Wp,
                                                   const __bf16* __restrict__ outT,
                                                   const float* __restrict__ x,
                                                   const float2* __restrict__ stats,
                                                   const float* __restrict__ g,
                                                   const float* __restrict__ be,
                                                   const float* __restrict__ bproj,
                                                   float* __restrict__ out) {
  const int idx = blockIdx.x;
  const int b = idx / 128;
  const int rem = idx % 128;
  const int mt = rem >> 5;        // 0..3
  const int nt = rem & 31;        // 0..31
  const int tid = threadIdx.x, wave = tid >> 5, lane = tid & 31;
  const int h = lane >> 4, lr = lane & 15;
  const __bf16* Bbase = outT + (size_t)b * HWN * CCH;

  v8f acc[4];
  gemm64x128(Wp, Bbase, mt * 64, nt * 128, tid, acc);

  const int m0 = mt * 64 + (wave >> 1) * 16;
  const int nbase = nt * 128 + (wave & 1) * 64;
  const float2 st = stats[b];
#pragma unroll
  for (int t = 0; t < 4; ++t) {
#pragma unroll
    for (int r = 0; r < 8; ++r) {
      const int o = m0 + r + 8 * h;
      const int hw = nbase + t * 16 + lr;
      const size_t gi = ((size_t)b * CCH + o) * HWN + hw;
      const float xn = (x[gi] - st.x) * st.y * g[o] + be[o];
      out[gi] = acc[t][r] + bproj[o] + xn;
    }
  }
}

// ---------------------------------------------------------------------------
extern "C" void kernel_launch(void* const* d_in, const int* in_sizes, int n_in,
                              void* d_out, int out_size, void* d_ws, size_t ws_size,
                              hipStream_t stream) {
  const float* x     = (const float*)d_in[0];
  const float* gamma = (const float*)d_in[1];
  const float* beta  = (const float*)d_in[2];
  const float* wqkv  = (const float*)d_in[3];
  const float* wproj = (const float*)d_in[4];
  const float* bproj = (const float*)d_in[5];
  float* out = (float*)d_out;

  char* ws = (char*)d_ws;
  float2* partials = (float2*)ws;                        // 4 KB
  float2* stats    = (float2*)(ws + ((size_t)8 << 10));  // 64 B
  __bf16* wqkvb    = (__bf16*)(ws + ((size_t)16 << 10)); // 384 KB
  __bf16* wprojb   = (__bf16*)(ws + ((size_t)416 << 10));// 128 KB
  __bf16* xnT      = (__bf16*)(ws + ((size_t)1 << 20));  // 16 MB [b][hw][c]
  __bf16* qT       = (__bf16*)(ws + ((size_t)17 << 20)); // 16 MB [b][h][i][c]
  __bf16* kf       = (__bf16*)(ws + ((size_t)33 << 20)); // 16 MB [b][c][hw]
  __bf16* vf       = (__bf16*)(ws + ((size_t)49 << 20)); // 16 MB [b][c][hw]
  __bf16* kpT      = (__bf16*)(ws + ((size_t)65 << 20)); // 4 MB  [b][h][j][c]
  __bf16* vp       = (__bf16*)(ws + ((size_t)70 << 20)); // 4 MB  [b][h][c][j]
  __bf16* outT     = (__bf16*)(ws + ((size_t)74 << 20)); // 16 MB [b][hw][c]

  gn_partial<<<BATCH * 64, 256, 0, stream>>>(x, partials);
  gn_final<<<BATCH, 64, 0, stream>>>(partials, stats);
  cvt_w<<<256, 256, 0, stream>>>(wqkv, wproj, wqkvb, wprojb);
  norm_xnT<<<BATCH * 16, 256, 0, stream>>>(x, stats, gamma, beta, xnT);
  qkv_gemm<<<BATCH * 12 * 32, 256, 0, stream>>>(wqkvb, xnT, qT, kf, vf);
  pool_kv<<<2048, 256, 0, stream>>>(kf, vf, kpT, vp);
  attn_kernel<<<BATCH * NH * 256, 256, 0, stream>>>(qT, kpT, vp, outT);
  proj_kernel<<<BATCH * 4 * 32, 256, 0, stream>>>(wprojb, outT, x, stats, gamma,
                                                  beta, bproj, out);
}
